// Resample_multi_channel_64630667870588
// MI455X (gfx1250) — matrix operand order repaired and checked by reading, not compiled, not run on hardware
//
#include <hip/hip_runtime.h>
#include <stdint.h>

// Problem constants (from reference): B=64, T=2048, C=32, OUT_T=2048
#define B_      64
#define T_      2048
#define C_      32
#define K_      (T_ * C_)        // 65536 reduction dim
#define N_      2048             // OUT_T
#define KSPLIT  4
#define KSLICE  (K_ / KSPLIT)    // 16384
#define CHUNK   64               // K staged per barrier (2 WMMA triples)
#define NCHUNK  (KSLICE / CHUNK) // 256
#define LDS_STRIDE 36            // u32 stride per n-row: 16B aligned, conflict-free b128

typedef __attribute__((ext_vector_type(16))) __bf16       v16bf;
typedef __attribute__((ext_vector_type(8)))  float        v8f;
typedef __attribute__((ext_vector_type(4)))  float        v4f;
typedef __attribute__((ext_vector_type(8)))  unsigned int v8u;

// pack two bf16 (truncated from f32) into one u32: low half = first value
static __device__ __forceinline__ unsigned pack_hi(float a, float b) {
  return (__builtin_bit_cast(unsigned, a) >> 16) |
         (__builtin_bit_cast(unsigned, b) & 0xffff0000u);
}
// residual after truncating to bf16 (exact in f32)
static __device__ __forceinline__ float trunc_res(float a) {
  return a - __builtin_bit_cast(float,
              __builtin_bit_cast(unsigned, a) & 0xffff0000u);
}
static __device__ __forceinline__ v16bf as_v16bf(uint4 a, uint4 b) {
  v8u u;
  u[0] = a.x; u[1] = a.y; u[2] = a.z; u[3] = a.w;
  u[4] = b.x; u[5] = b.y; u[6] = b.z; u[7] = b.w;
  return __builtin_bit_cast(v16bf, u);
}

// ---------------------------------------------------------------------------
// Kernel 1: split X (fp32) into packed bf16 hi/lo pair arrays.
// Xh[i] packs {bf16(X[2i]) , bf16(X[2i+1])} (even K in low half) — exactly the
// per-lane pair order of the 16-bit WMMA A fragment.
// ---------------------------------------------------------------------------
__global__ void __launch_bounds__(256)
prep_split_x(const float* __restrict__ X,
             unsigned* __restrict__ Xh, unsigned* __restrict__ Xl) {
  int i = blockIdx.x * blockDim.x + threadIdx.x;   // 0 .. 524287 (each = 4 pairs)
  const v4f* x4 = (const v4f*)X;
  v4f f0 = x4[2 * i];
  v4f f1 = x4[2 * i + 1];
  uint4 h, l;
  h.x = pack_hi(f0.x, f0.y);  h.y = pack_hi(f0.z, f0.w);
  h.z = pack_hi(f1.x, f1.y);  h.w = pack_hi(f1.z, f1.w);
  l.x = pack_hi(trunc_res(f0.x), trunc_res(f0.y));
  l.y = pack_hi(trunc_res(f0.z), trunc_res(f0.w));
  l.z = pack_hi(trunc_res(f1.x), trunc_res(f1.y));
  l.w = pack_hi(trunc_res(f1.z), trunc_res(f1.w));
  ((uint4*)Xh)[i] = h;
  ((uint4*)Xl)[i] = l;
}

// ---------------------------------------------------------------------------
// Kernel 2: deformation GEMM via split-bf16 WMMA.
//   D = X(64 x 65536) @ W(65536 x 2048), K split 4 ways into partial buffers.
// Block = 128 threads (4 waves); wave w owns M-tile rows [16w, 16w+16).
// Per K=64 chunk the block stages the 64x16 W tile into LDS as packed bf16
// hi/lo K-pairs in [n][kpair] order (B fragment = 2 x ds_load_b128).
// W is streamed nontemporal: each byte of W is read exactly once on the chip.
// ---------------------------------------------------------------------------
__global__ void __launch_bounds__(128)
locnet_gemm(const float* __restrict__ W,
            const unsigned* __restrict__ Xh, const unsigned* __restrict__ Xl,
            float* __restrict__ Part) {
  __shared__ unsigned ldsb[2][2][16 * LDS_STRIDE];  // [buf][hi/lo][n*36+kp] 9KB

  const int nTile  = blockIdx.x;       // 0..127
  const int slice  = blockIdx.y;       // 0..3
  const int nBase  = nTile * 16;
  const int kStart = slice * KSLICE;

  const int tid  = threadIdx.x;
  const int wave = tid >> 5;
  const int lane = tid & 31;
  const int h    = lane >> 4;          // half-wave select
  const int n16  = lane & 15;
  const int mBase = wave * 16;
  const int m     = mBase + n16;       // this lane's A row

  // staging role: thread owns K-pair kp (0..31), column quad n0..n0+3
  const int kp = tid >> 2;             // 0..31
  const int n0 = (tid & 3) * 4;

  auto stage = [&](int buf, int kc) {
    size_t k0 = (size_t)(kStart + kc + 2 * kp);
    const float* r0p = W + k0 * N_ + nBase + n0;
    v4f r0 = __builtin_nontemporal_load((const v4f*)r0p);        // row k0 (even K)
    v4f r1 = __builtin_nontemporal_load((const v4f*)(r0p + N_)); // row k0+1 (odd K)
    // prefetch the W stream two chunks ahead
    if (kc + 2 * CHUNK < KSLICE)
      __builtin_prefetch(r0p + (size_t)2 * CHUNK * N_, 0, 1);
    unsigned* bh = &ldsb[buf][0][n0 * LDS_STRIDE + kp];
    unsigned* bl = &ldsb[buf][1][n0 * LDS_STRIDE + kp];
    bh[0 * LDS_STRIDE] = pack_hi(r0.x, r1.x);
    bh[1 * LDS_STRIDE] = pack_hi(r0.y, r1.y);
    bh[2 * LDS_STRIDE] = pack_hi(r0.z, r1.z);
    bh[3 * LDS_STRIDE] = pack_hi(r0.w, r1.w);
    bl[0 * LDS_STRIDE] = pack_hi(trunc_res(r0.x), trunc_res(r1.x));
    bl[1 * LDS_STRIDE] = pack_hi(trunc_res(r0.y), trunc_res(r1.y));
    bl[2 * LDS_STRIDE] = pack_hi(trunc_res(r0.z), trunc_res(r1.z));
    bl[3 * LDS_STRIDE] = pack_hi(trunc_res(r0.w), trunc_res(r1.w));
  };

  const uint4* xh4 = (const uint4*)(Xh + (size_t)m * (K_ / 2));
  const uint4* xl4 = (const uint4*)(Xl + (size_t)m * (K_ / 2));

  v8f acc = {};

  stage(0, 0);
  __syncthreads();

  for (int ch = 0; ch < NCHUNK; ++ch) {
    const int buf = ch & 1;
    if (ch + 1 < NCHUNK) stage(buf ^ 1, (ch + 1) * CHUNK);

#pragma unroll
    for (int s = 0; s < 2; ++s) {       // two K=32 sub-steps per staged chunk
      // A fragment: lane holds K = [kc+8h, +8) then [kc+16+8h, +8)
      const int q = ((kStart + ch * CHUNK + 32 * s) >> 3) + h;  // uint4 idx in row
      uint4 a_h0 = xh4[q], a_h1 = xh4[q + 2];
      uint4 a_l0 = xl4[q], a_l1 = xl4[q + 2];

      // B fragment: VGPR v <- kpair (16s + 4h + v) for v<4, (16s + 8 + 4h + v-4) for v>=4 (matches A)
      const unsigned* bhp = &ldsb[buf][0][n16 * LDS_STRIDE + 16 * s + 4 * h];
      const unsigned* blp = &ldsb[buf][1][n16 * LDS_STRIDE + 16 * s + 4 * h];
      uint4 b_h0 = *(const uint4*)(bhp);
      uint4 b_h1 = *(const uint4*)(bhp + 8);
      uint4 b_l0 = *(const uint4*)(blp);
      uint4 b_l1 = *(const uint4*)(blp + 8);

      v16bf Ah = as_v16bf(a_h0, a_h1), Al = as_v16bf(a_l0, a_l1);
      v16bf Bh = as_v16bf(b_h0, b_h1), Bl = as_v16bf(b_l0, b_l1);

      // split-fp32 product: ah*bh + al*bh + ah*bl  (al*bl ~ 2^-16, dropped)
      acc = __builtin_amdgcn_wmma_f32_16x16x32_bf16(false, Ah, false, Bh,
                                                    (short)0, acc, false, false);
      acc = __builtin_amdgcn_wmma_f32_16x16x32_bf16(false, Al, false, Bh,
                                                    (short)0, acc, false, false);
      acc = __builtin_amdgcn_wmma_f32_16x16x32_bf16(false, Ah, false, Bl,
                                                    (short)0, acc, false, false);
    }
    __syncthreads();
  }

  // C/D layout: VGPR v -> row mBase + v + 8h, col nBase + n16
  float* P = Part + (size_t)slice * (B_ * N_);
#pragma unroll
  for (int v = 0; v < 8; ++v) {
    int row = mBase + v + 8 * h;
    P[(size_t)row * N_ + nBase + n16] = acc[v];
  }
}

// ---------------------------------------------------------------------------
// Kernel 3: sum K-split partials + bias -> sample coords -> clamped lerp.
// ---------------------------------------------------------------------------
__global__ void __launch_bounds__(256)
resample(const float* __restrict__ X, const float* __restrict__ bias,
         const float* __restrict__ Part, float* __restrict__ out) {
  int idx = blockIdx.x * blockDim.x + threadIdx.x;  // 0 .. B_*N_-1
  int b  = idx >> 11;
  int ot = idx & (N_ - 1);
  size_t di = (size_t)b * N_ + ot;
  float def = bias[ot]
            + Part[di]
            + Part[di + (size_t)1 * B_ * N_]
            + Part[di + (size_t)2 * B_ * N_]
            + Part[di + (size_t)3 * B_ * N_];
  float g = (float)ot * ((float)(T_ - 1) / (float)(N_ - 1));
  float x = g + def;
  float xf = floorf(x);
  int x0 = (int)xf;
  int x1 = x0 + 1;
  int x0c = min(max(x0, 0), T_ - 1);
  int x1c = min(max(x1, 0), T_ - 1);
  float w0 = (float)x1c - x;   // weights from clipped coords (faithful)
  float w1 = x - (float)x0c;

  const v4f* v0 = (const v4f*)(X + ((size_t)b * T_ + x0c) * C_);
  const v4f* v1 = (const v4f*)(X + ((size_t)b * T_ + x1c) * C_);
  v4f* o = (v4f*)(out + (size_t)idx * C_);
#pragma unroll
  for (int c4 = 0; c4 < C_ / 4; ++c4) {
    v4f a = v0[c4], bb = v1[c4];
    v4f r = w0 * a + w1 * bb;
    __builtin_nontemporal_store(r, &o[c4]);   // out written once, never re-read
  }
}

// ---------------------------------------------------------------------------
extern "C" void kernel_launch(void* const* d_in, const int* in_sizes, int n_in,
                              void* d_out, int out_size, void* d_ws, size_t ws_size,
                              hipStream_t stream) {
  const float* X    = (const float*)d_in[0];   // (64, 2048, 32) f32
  const float* W    = (const float*)d_in[1];   // (65536, 2048) f32
  const float* bloc = (const float*)d_in[2];   // (2048,) f32

  // workspace: Xh (8MB) | Xl (8MB) | partials (4 * 64*2048 f32 = 2MB)
  unsigned* Xh = (unsigned*)d_ws;
  unsigned* Xl = Xh + (size_t)B_ * (K_ / 2);
  float* Part  = (float*)(Xl + (size_t)B_ * (K_ / 2));
  float* out   = (float*)d_out;

  prep_split_x<<<2048, 256, 0, stream>>>(X, Xh, Xl);

  dim3 grid(N_ / 16, KSPLIT);                 // 128 N-tiles x 4 K-slices
  locnet_gemm<<<grid, 128, 0, stream>>>(W, Xh, Xl, Part);

  resample<<<(B_ * N_) / 256, 256, 0, stream>>>(X, bloc, Part, out);
}
